// SamplingEncAttention_25280177504718
// MI455X (gfx1250) — matrix operand order repaired and checked by reading, not compiled, not run on hardware
//
#include <hip/hip_runtime.h>
#include <stdint.h>

// ---------------- problem constants ----------------
#define N_B    2
#define C_DIM  512
#define M_H    8
#define FL_L   4
#define CM_D   64
#define S_TOT  13294          // 100*100 + 50*50 + 25*25 + 13*13
#define SP_TOT 13312          // padded to multiple of 16
#define OW_ROWS 96            // 64 offset rows + 32 weight rows

typedef __attribute__((ext_vector_type(16))) __bf16 v16bf;
typedef __attribute__((ext_vector_type(8)))  float  v8f;

__device__ __constant__ int d_lvl_off[5] = {0, 10000, 12500, 13125, 13294};
__device__ __constant__ int d_lvl_hw[4]  = {100, 50, 25, 13};   // square levels

__device__ __forceinline__ __bf16 f2bf(float f) {
    unsigned u = __float_as_uint(f);
    u += 0x7fffu + ((u >> 16) & 1u);            // round-to-nearest-even
    unsigned short h = (unsigned short)(u >> 16);
    return __builtin_bit_cast(__bf16, h);
}

// ---------------- prep kernels ----------------
__global__ void cvt_bf16_kernel(const float* __restrict__ src,
                                __bf16* __restrict__ dst, int n) {
    int i = blockIdx.x * blockDim.x + threadIdx.x;
    if (i < n) dst[i] = f2bf(src[i]);
}

__global__ void bias_cat_kernel(const float* __restrict__ bloc,
                                const float* __restrict__ bw,
                                float* __restrict__ bcat) {
    int i = threadIdx.x;
    if (i < 64)            bcat[i] = bloc[i];
    else if (i < OW_ROWS)  bcat[i] = bw[i - 64];
}

// Transpose + convert: xb[n][s][c] = bf16(x), xpb[n][s][c] = bf16(mask?0:x+pos).
// One thread per (n, s): coalesced reads across lanes (consecutive s per c),
// each thread writes its full 1KB row with b128 stores (complete lines).
__global__ __launch_bounds__(256) void prep_x_kernel(
        const float* __restrict__ x, const float* __restrict__ pos,
        const uint8_t* __restrict__ kpm,
        __bf16* __restrict__ xb, __bf16* __restrict__ xpb) {
    int t = blockIdx.x * blockDim.x + threadIdx.x;      // over N*SP
    if (t >= N_B * SP_TOT) return;
    int n  = t / SP_TOT;
    int sp = t % SP_TOT;
    bool inb = sp < S_TOT;
    uint8_t mv = inb ? kpm[(size_t)n * S_TOT + sp] : 1;
    __bf16* xrow = xb  + (size_t)t * C_DIM;
    __bf16* prow = xpb + (size_t)t * C_DIM;
    const float* xc = x   + (size_t)n * C_DIM * S_TOT + sp;
    const float* pc = pos + (size_t)n * C_DIM * S_TOT + sp;
    for (int c0 = 0; c0 < C_DIM; c0 += 8) {
        union { __bf16 h[8]; uint4 u; } a, b;
#pragma unroll
        for (int j = 0; j < 8; ++j) {
            float xv = inb ? xc[(size_t)(c0 + j) * S_TOT] : 0.f;
            float pv = inb ? pc[(size_t)(c0 + j) * S_TOT] : 0.f;
            a.h[j] = f2bf(xv);
            b.h[j] = f2bf(mv ? 0.f : xv + pv);
        }
        *(uint4*)(xrow + c0) = a.u;
        *(uint4*)(prow + c0) = b.u;
    }
}

// ---------------- WMMA GEMM (transposed formulation, o-blocked) ----------------
// D^T tiles: Y^T[s, o] = sum_k Xs[n][s][k] * Wb[o][k]
// Each wave computes a 16s x (16*OT)o strip: the A fragment (X rows, 2x b128)
// is reused across OT WMMAs; each B fragment (W rows) is 2x contiguous b128.
// FINAL=false: store s-major out[n][s][O].  FINAL=true: b64 stores into d_out[n][o][S]*scale.
template<int OT, bool MASKED, bool FINAL>
__global__ __launch_bounds__(256) void wmma_gemm_kernel(
        const __bf16* __restrict__ Xs,    // N x SP x C  (s-major bf16)
        const __bf16* __restrict__ Wb,    // O x C bf16 row-major
        const float*  __restrict__ bias,  // O
        const uint8_t* __restrict__ mask, // N x S_TOT (or null)
        const float*  __restrict__ scalev,// O (or null)
        float* __restrict__ out, int O) {
    const int n      = blockIdx.z;
    const int sbase  = blockIdx.x * 16;
    const int wave   = threadIdx.x >> 5;
    const int lane   = threadIdx.x & 31;
    const int wpb    = blockDim.x >> 5;
    const int obase0 = (blockIdx.y * wpb + wave) * (16 * OT);
    const int half   = lane >> 4;
    const int l16    = lane & 15;
    const int kh     = half * 8;
    const int klo    = half * 16;

    const __bf16* arow = Xs + ((size_t)n * SP_TOT + sbase + l16) * C_DIM;
    const __bf16* brow = Wb + (size_t)(obase0 + l16) * C_DIM;   // tile j at +j*16*C_DIM

    v8f acc[OT];
#pragma unroll
    for (int j = 0; j < OT; ++j) acc[j] = (v8f){};

#pragma unroll 2
    for (int k0 = 0; k0 < C_DIM; k0 += 32) {
        union { uint4 u4[2]; v16bf v; } af;
        af.u4[0] = *(const uint4*)(arow + k0 + kh);
        af.u4[1] = *(const uint4*)(arow + k0 + 16 + kh);
        if (k0 + 32 < C_DIM)
            __builtin_prefetch(arow + k0 + 32, 0, 3);   // WGP-scope global_prefetch
#pragma unroll
        for (int j = 0; j < OT; ++j) {
            union { uint4 u4[2]; v16bf v; } bf;
            const __bf16* bj = brow + (size_t)j * 16 * C_DIM + k0 + kh;
            bf.u4[0] = *(const uint4*)(bj);
            bf.u4[1] = *(const uint4*)(bj + 16);
            acc[j] = __builtin_amdgcn_wmma_f32_16x16x32_bf16(
                /*neg_a=*/false, af.v, /*neg_b=*/false, bf.v,
                /*c_mod=*/(short)0, acc[j], /*reuse_a=*/false, /*reuse_b=*/false);
        }
    }

    // D^T epilogue: VGPR v -> s = sbase + 8*half + v, o = obase + (lane&15)
#pragma unroll
    for (int j = 0; j < OT; ++j) {
        const int   o   = obase0 + j * 16 + l16;
        const float bo_ = bias[o];
        const float sc_ = FINAL ? scalev[o] : 1.f;
        if (FINAL) {
            // lane's 8 results are contiguous in s at fixed o -> 4x b64 stores
            const int s0 = sbase + half * 8;
            float r[8];
#pragma unroll
            for (int v = 0; v < 8; ++v) r[v] = (acc[j][v] + bo_) * sc_;
            float* dst = out + ((size_t)n * C_DIM + o) * S_TOT + s0;
            if (s0 + 7 < S_TOT) {
#pragma unroll
                for (int v = 0; v < 8; v += 2)
                    *(float2*)(dst + v) = make_float2(r[v], r[v + 1]);
            } else {
#pragma unroll
                for (int v = 0; v < 8; ++v)
                    if (s0 + v < S_TOT) dst[v] = r[v];
            }
        } else {
#pragma unroll
            for (int v = 0; v < 8; ++v) {
                int s = sbase + half * 8 + v;
                float val = acc[j][v] + bo_;
                if (MASKED) {
                    uint8_t mv = (s < S_TOT) ? mask[(size_t)n * S_TOT + s] : 1;
                    if (mv) val = 0.f;
                }
                out[((size_t)n * SP_TOT + s) * O + o] = val;  // 64B runs across lanes
            }
        }
    }
}

// ---------------- deformable sampling ----------------
// block: (64 channels, 4 heads); grid: (S, M/4, N). All buffers s-major.
__global__ __launch_bounds__(256) void sampling_kernel(
        const float* __restrict__ offw,   // N x SP x 96
        const float* __restrict__ value,  // N x SP x C
        const float* __restrict__ vsizes, // N x FL x 2  (w,h)
        const float* __restrict__ vscales,// N x FL x 2
        __bf16* __restrict__ midb) {      // N x SP x C
    const int s  = blockIdx.x;
    const int m  = blockIdx.y * blockDim.y + threadIdx.y;
    const int n  = blockIdx.z;
    const int cc = threadIdx.x;

    int lvl = 0;
    if (s >= d_lvl_off[1]) lvl = 1;
    if (s >= d_lvl_off[2]) lvl = 2;
    if (s >= d_lvl_off[3]) lvl = 3;
    const int l  = s - d_lvl_off[lvl];
    const int Wl = d_lvl_hw[lvl];
    const float qx = (float)(l % Wl) + 0.5f;
    const float qy = (float)(l / Wl) + 0.5f;

    const float vw  = vsizes[((size_t)n * FL_L + lvl) * 2 + 0];
    const float vh  = vsizes[((size_t)n * FL_L + lvl) * 2 + 1];
    const float scx = 2.f * vscales[((size_t)n * FL_L + lvl) * 2 + 0] / vw;
    const float scy = 2.f * vscales[((size_t)n * FL_L + lvl) * 2 + 1] / vh;
    const float bx  = qx * scx - 1.f;
    const float by  = qy * scy - 1.f;

    const float* orow = offw + ((size_t)n * SP_TOT + s) * OW_ROWS;

    // softmax over 4 levels (one cache line per query)
    float lg[FL_L], mx = -3.0e38f;
#pragma unroll
    for (int f = 0; f < FL_L; ++f) {
        lg[f] = orow[64 + m * FL_L + f];
        mx = fmaxf(mx, lg[f]);
    }
    float ssum = 0.f;
#pragma unroll
    for (int f = 0; f < FL_L; ++f) { lg[f] = __expf(lg[f] - mx); ssum += lg[f]; }
    const float inv = 1.f / ssum;

    const float* vb = value + ((size_t)n * SP_TOT) * C_DIM + m * CM_D + cc;

    float accv = 0.f;
#pragma unroll
    for (int f = 0; f < FL_L; ++f) {
        const float ox = orow[m * (FL_L * 2) + f * 2 + 0];
        const float oy = orow[m * (FL_L * 2) + f * 2 + 1];
        const float gx = ox * scx + bx;
        const float gy = oy * scy + by;
        const int Wf = d_lvl_hw[f];
        const int Hf = Wf;
        const float xi = (gx + 1.f) * (Wf * 0.5f) - 0.5f;
        const float yi = (gy + 1.f) * (Hf * 0.5f) - 0.5f;
        const float x0f = floorf(xi), y0f = floorf(yi);
        const int   x0 = (int)x0f,   y0 = (int)y0f;
        const float wx1 = xi - x0f,  wy1 = yi - y0f;
        const float wx0 = 1.f - wx1, wy0 = 1.f - wy1;
        const int lo = d_lvl_off[f];
        auto tap = [&](int xx, int yy) -> float {
            if (xx < 0 || xx >= Wf || yy < 0 || yy >= Hf) return 0.f;
            return vb[(size_t)(lo + yy * Wf + xx) * C_DIM];   // 256B coalesced over cc
        };
        float sv = tap(x0, y0)         * wx0 * wy0
                 + tap(x0 + 1, y0)     * wx1 * wy0
                 + tap(x0, y0 + 1)     * wx0 * wy1
                 + tap(x0 + 1, y0 + 1) * wx1 * wy1;
        accv += sv * (lg[f] * inv);
    }
    midb[((size_t)n * SP_TOT + s) * C_DIM + m * CM_D + cc] = f2bf(accv);
}

// ---------------- host orchestration ----------------
extern "C" void kernel_launch(void* const* d_in, const int* in_sizes, int n_in,
                              void* d_out, int out_size, void* d_ws, size_t ws_size,
                              hipStream_t stream) {
    const float*   x       = (const float*)d_in[0];
    const float*   pos     = (const float*)d_in[1];
    const uint8_t* kpm     = (const uint8_t*)d_in[2];
    const float*   vsizes  = (const float*)d_in[3];
    const float*   vscales = (const float*)d_in[4];
    const float*   Wv      = (const float*)d_in[5];
    const float*   bv      = (const float*)d_in[6];
    const float*   Wloc    = (const float*)d_in[7];
    const float*   bloc    = (const float*)d_in[8];
    const float*   Ww      = (const float*)d_in[9];
    const float*   bw      = (const float*)d_in[10];
    const float*   Wo      = (const float*)d_in[11];
    const float*   bo      = (const float*)d_in[12];
    const float*   scale   = (const float*)d_in[13];
    float*         out     = (float*)d_out;

    // workspace carve-up
    char*  ws  = (char*)d_ws;
    size_t off = 0;
    auto carve = [&](size_t bytes) { void* p = ws + off; off = (off + bytes + 255) & ~(size_t)255; return p; };
    __bf16* Wvb   = (__bf16*)carve((size_t)C_DIM * C_DIM * 2);
    __bf16* Wob   = (__bf16*)carve((size_t)C_DIM * C_DIM * 2);
    __bf16* Wcatb = (__bf16*)carve((size_t)OW_ROWS * C_DIM * 2);
    float*  bcat  = (float*) carve((size_t)OW_ROWS * 4);
    __bf16* xb    = (__bf16*)carve((size_t)N_B * SP_TOT * C_DIM * 2);
    __bf16* xpb   = (__bf16*)carve((size_t)N_B * SP_TOT * C_DIM * 2);
    float*  value = (float*) carve((size_t)N_B * SP_TOT * C_DIM * 4);
    __bf16* midb  = (__bf16*)carve((size_t)N_B * SP_TOT * C_DIM * 2);
    float*  offw  = (float*) carve((size_t)N_B * SP_TOT * OW_ROWS * 4);
    (void)ws_size; (void)in_sizes; (void)n_in; (void)out_size;

    // 1) weight conversions + bias concat
    {
        int n1 = C_DIM * C_DIM;
        cvt_bf16_kernel<<<(n1 + 255) / 256, 256, 0, stream>>>(Wv, Wvb, n1);
        cvt_bf16_kernel<<<(n1 + 255) / 256, 256, 0, stream>>>(Wo, Wob, n1);
        int n2 = 64 * C_DIM, n3 = 32 * C_DIM;
        cvt_bf16_kernel<<<(n2 + 255) / 256, 256, 0, stream>>>(Wloc, Wcatb, n2);
        cvt_bf16_kernel<<<(n3 + 255) / 256, 256, 0, stream>>>(Ww, Wcatb + n2, n3);
        bias_cat_kernel<<<1, 128, 0, stream>>>(bloc, bw, bcat);
    }
    // 2) x / x_pos -> bf16, transposed to s-major, padded rows zeroed
    {
        int total = N_B * SP_TOT;
        prep_x_kernel<<<(total + 255) / 256, 256, 0, stream>>>(x, pos, kpm, xb, xpb);
    }
    // 3) value[n][s][c] = Wv*x + bv, masked   (8 waves x 64o = 512)
    {
        dim3 g(SP_TOT / 16, 1, N_B);
        wmma_gemm_kernel<4, true, false><<<g, 256, 0, stream>>>(
            xb, Wvb, bv, kpm, nullptr, value, C_DIM);
    }
    // 4) offw[n][s][96] = [Wloc;Ww]*x_pos + [bloc;bw]   (6 o-tiles -> 6 waves)
    {
        dim3 g(SP_TOT / 16, 1, N_B);
        wmma_gemm_kernel<1, false, false><<<g, 192, 0, stream>>>(
            xpb, Wcatb, bcat, nullptr, nullptr, offw, OW_ROWS);
    }
    // 5) deformable sampling -> midb[n][s][c] (bf16)
    {
        dim3 g(S_TOT, M_H / 4, N_B);
        dim3 b(CM_D, 4);
        sampling_kernel<<<g, b, 0, stream>>>(offw, value, vsizes, vscales, midb);
    }
    // 6) d_out[n][o][s] = (Wo*mid + bo) * scale   (8 waves x 64o = 512)
    {
        dim3 g(SP_TOT / 16, 1, N_B);
        wmma_gemm_kernel<4, false, true><<<g, 256, 0, stream>>>(
            midb, Wob, bo, nullptr, scale, out, C_DIM);
    }
}
